// HyperbolicRGCNLayer_26680336843171
// MI455X (gfx1250) — matrix operand order repaired and checked by reading, not compiled, not run on hardware
//
#include <hip/hip_runtime.h>
#include <math.h>

// ---------------------------------------------------------------------------
// HyperbolicRGCN layer for MI455X (gfx1250, wave32).
//   k0: zero agg + build bf16 transposed loop_weight
//   k1: log-map (tangent f32 + bf16 copy + radius)        [1 wave / node]
//   k2: per-edge block-diag matvec + radius weight + f32 atomic scatter
//   k3: self-loop GEMM via v_wmma_f32_16x16x32_bf16, C seeded with agg*norm
//   k4: exp-map -> d_out
// ---------------------------------------------------------------------------

typedef __attribute__((ext_vector_type(16))) __bf16 v16bf;
typedef __attribute__((ext_vector_type(8)))  __bf16 v8bf;
typedef __attribute__((ext_vector_type(8)))  float  v8f;

#define SQRT_C        0.1f
#define EPS_F         1e-10f
#define MAX_TANH_ARG  (1.0f - 1e-5f)
#define IN_FEAT       256
#define OUT_FEAT      256

union FragBF { v16bf v; v8bf h[2]; };
union Oct    { v8bf v; __bf16 e[8]; };

// ---------------- k0: zero agg + transpose/convert loop_weight -------------
__global__ __launch_bounds__(256)
void k0_init(const float* __restrict__ lw, __bf16* __restrict__ wt,
             float* __restrict__ agg, int aggN) {
    int i = blockIdx.x * 256 + threadIdx.x;
    if (i < aggN) agg[i] = 0.0f;
    if (i < 256 * 256) {
        int o = i & 255, k = i >> 8;          // wt[o][k] = lw[k][o]
        wt[o * 256 + k] = (__bf16)lw[k * 256 + o];
    }
}

// ---------------- k1: log map at origin ------------------------------------
__global__ __launch_bounds__(256)
void k1_tangent(const float* __restrict__ h, float* __restrict__ xt,
                __bf16* __restrict__ xtb, float* __restrict__ radius, int n) {
    int wid  = (blockIdx.x * 256 + threadIdx.x) >> 5;
    int lane = threadIdx.x & 31;
    if (wid >= n) return;
    const float4* hp = (const float4*)(h + wid * IN_FEAT + lane * 8);
    float4 a = hp[0], b = hp[1];
    float ss = a.x*a.x + a.y*a.y + a.z*a.z + a.w*a.w
             + b.x*b.x + b.y*b.y + b.z*b.z + b.w*b.w;
    #pragma unroll
    for (int m = 16; m >= 1; m >>= 1) ss += __shfl_xor(ss, m, 32);
    float nrm = fmaxf(sqrtf(ss), EPS_F);
    float arg = fminf(fmaxf(SQRT_C * nrm, EPS_F), MAX_TANH_ARG);
    float ath = 0.5f * __logf((1.0f + arg) / (1.0f - arg));   // artanh
    float sc  = ath / (SQRT_C * nrm);
    float4 o0 = make_float4(a.x*sc, a.y*sc, a.z*sc, a.w*sc);
    float4 o1 = make_float4(b.x*sc, b.y*sc, b.z*sc, b.w*sc);
    float4* op = (float4*)(xt + wid * IN_FEAT + lane * 8);
    op[0] = o0; op[1] = o1;
    Oct q;
    q.e[0]=(__bf16)o0.x; q.e[1]=(__bf16)o0.y; q.e[2]=(__bf16)o0.z; q.e[3]=(__bf16)o0.w;
    q.e[4]=(__bf16)o1.x; q.e[5]=(__bf16)o1.y; q.e[6]=(__bf16)o1.z; q.e[7]=(__bf16)o1.w;
    *(v8bf*)(xtb + wid * IN_FEAT + lane * 8) = q.v;
    if (lane == 0) radius[wid] = (2.0f / SQRT_C) * ath;
}

// ---------------- k2: edge messages + atomic scatter -----------------------
// One wave per edge. Lane l handles bases 2l,2l+1 -> outputs [8l,8l+8).
__global__ __launch_bounds__(256)
void k2_edges(const float* __restrict__ xt, const float* __restrict__ radius,
              const float* __restrict__ W, const int* __restrict__ src,
              const int* __restrict__ dst, const int* __restrict__ etype,
              float* __restrict__ agg, int e) {
    int wid  = (blockIdx.x * 256 + threadIdx.x) >> 5;
    int lane = threadIdx.x & 31;
    if (wid >= e) return;
    int s = src[wid], d = dst[wid], t = etype[wid];
    float rw = __expf(-fabsf(radius[s] - radius[d]));

    const float4* xp = (const float4*)(xt + s * IN_FEAT + lane * 8);
    float4 x0 = xp[0], x1 = xp[1];
    const float4* wp = (const float4*)(W + t * 1024 + lane * 32);
    float4 w0 = wp[0], w1 = wp[1], w2 = wp[2], w3 = wp[3];   // base 2l
    float4 w4 = wp[4], w5 = wp[5], w6 = wp[6], w7 = wp[7];   // base 2l+1

    float o[8];
    o[0] = x0.x*w0.x + x0.y*w1.x + x0.z*w2.x + x0.w*w3.x;
    o[1] = x0.x*w0.y + x0.y*w1.y + x0.z*w2.y + x0.w*w3.y;
    o[2] = x0.x*w0.z + x0.y*w1.z + x0.z*w2.z + x0.w*w3.z;
    o[3] = x0.x*w0.w + x0.y*w1.w + x0.z*w2.w + x0.w*w3.w;
    o[4] = x1.x*w4.x + x1.y*w5.x + x1.z*w6.x + x1.w*w7.x;
    o[5] = x1.x*w4.y + x1.y*w5.y + x1.z*w6.y + x1.w*w7.y;
    o[6] = x1.x*w4.z + x1.y*w5.z + x1.z*w6.z + x1.w*w7.z;
    o[7] = x1.x*w4.w + x1.y*w5.w + x1.z*w6.w + x1.w*w7.w;

    float* ap = agg + d * OUT_FEAT + lane * 8;
    #pragma unroll
    for (int j = 0; j < 8; ++j) unsafeAtomicAdd(ap + j, o[j] * rw);
}

// ---------------- k3: self-loop GEMM (WMMA bf16, f32 accum) ----------------
// One wave computes a 16x16 tile of  agg*norm + Xt(bf16) @ loop_weight(bf16).
__global__ __launch_bounds__(256)
void k3_gemm(const __bf16* __restrict__ A,   // [n][256] bf16 tangent
             const __bf16* __restrict__ Bt,  // [256][256] bf16, Bt[out][in]
             const float* __restrict__ nv,   // [n] norm
             float* __restrict__ agg, int n) {
    int lane = threadIdx.x & 31;
    int wave = threadIdx.x >> 5;
    int tile = blockIdx.x * 8 + wave;
    int mt = tile >> 4, nt = tile & 15;
    int row0 = mt * 16;
    if (row0 >= n) return;
    int col0 = nt * 16;
    int lm = lane & 15, lh = lane >> 4;
    int ccol = col0 + lm;

    // Seed C with agg * norm  (C/D layout: vgpr r -> M = r + lh*8, N = lm)
    v8f c;
    #pragma unroll
    for (int r = 0; r < 8; ++r) {
        int row = row0 + r + lh * 8;
        c[r] = agg[row * OUT_FEAT + ccol] * nv[row];
    }

    const __bf16* arow = A  + (size_t)(row0 + lm) * IN_FEAT;  // A row for lane
    const __bf16* bcol = Bt + (size_t)ccol * IN_FEAT;         // B column (contig K)

    #pragma unroll
    for (int kk = 0; kk < IN_FEAT; kk += 32) {
        FragBF fa, fb;
        // A 16x32 bf16 layout: lanes0-15 K=[kk,kk+8)+[kk+16,kk+24); lanes16-31 +8
        fa.h[0] = *(const v8bf*)(arow + kk + lh * 8);
        fa.h[1] = *(const v8bf*)(arow + kk + 16 + lh * 8);
        // B 32x16 layout: same K order as A: lanes0-15 K=[kk,kk+8)+[kk+16,kk+24); lanes16-31 +8, col = lm
        fb.h[0] = *(const v8bf*)(bcol + kk + lh * 8);
        fb.h[1] = *(const v8bf*)(bcol + kk + 16 + lh * 8);
        c = __builtin_amdgcn_wmma_f32_16x16x32_bf16(false, fa.v, false, fb.v,
                                                    (short)0, c, false, false);
    }

    #pragma unroll
    for (int r = 0; r < 8; ++r) {
        int row = row0 + r + lh * 8;
        agg[row * OUT_FEAT + ccol] = c[r];     // h_new, in place
    }
}

// ---------------- k4: exp map at origin -> output --------------------------
__global__ __launch_bounds__(256)
void k4_expmap(const float* __restrict__ hn, float* __restrict__ out, int n) {
    int wid  = (blockIdx.x * 256 + threadIdx.x) >> 5;
    int lane = threadIdx.x & 31;
    if (wid >= n) return;
    const float4* hp = (const float4*)(hn + wid * OUT_FEAT + lane * 8);
    float4 a = hp[0], b = hp[1];
    float ss = a.x*a.x + a.y*a.y + a.z*a.z + a.w*a.w
             + b.x*b.x + b.y*b.y + b.z*b.z + b.w*b.w;
    #pragma unroll
    for (int m = 16; m >= 1; m >>= 1) ss += __shfl_xor(ss, m, 32);
    float nrm = fmaxf(sqrtf(ss), EPS_F);
    float sn  = SQRT_C * nrm;
    float sc  = tanhf(sn) / sn;
    float4* op = (float4*)(out + wid * OUT_FEAT + lane * 8);
    op[0] = make_float4(a.x*sc, a.y*sc, a.z*sc, a.w*sc);
    op[1] = make_float4(b.x*sc, b.y*sc, b.z*sc, b.w*sc);
}

// ---------------------------------------------------------------------------
extern "C" void kernel_launch(void* const* d_in, const int* in_sizes, int n_in,
                              void* d_out, int out_size, void* d_ws, size_t ws_size,
                              hipStream_t stream) {
    const float* h_hyper = (const float*)d_in[0];   // [N,256]
    const float* weight  = (const float*)d_in[1];   // [256,1024]
    const float* loop_w  = (const float*)d_in[2];   // [256,256]
    const float* normv   = (const float*)d_in[3];   // [N,1]
    const int*   src     = (const int*)d_in[4];     // [E]
    const int*   dst     = (const int*)d_in[5];     // [E]
    const int*   etype   = (const int*)d_in[6];     // [E]
    float* out = (float*)d_out;

    const int n = in_sizes[3];          // 10000 (divisible by 16)
    const int e = in_sizes[4];          // 320000

    // workspace layout (256B aligned)
    char* w = (char*)d_ws;
    size_t off = 0;
    auto take = [&](size_t bytes) { char* p = w + off;
                                    off = (off + bytes + 255) & ~(size_t)255; return p; };
    float*   xt     = (float*)  take((size_t)n * IN_FEAT * 4);   // tangent f32
    __bf16*  xtb    = (__bf16*) take((size_t)n * IN_FEAT * 2);   // tangent bf16
    float*   radius = (float*)  take((size_t)n * 4);
    float*   agg    = (float*)  take((size_t)n * OUT_FEAT * 4);  // agg / h_new
    __bf16*  wt     = (__bf16*) take((size_t)256 * 256 * 2);     // loop_w^T bf16
    (void)ws_size;

    const int aggN = n * OUT_FEAT;
    k0_init   <<<(aggN + 255) / 256, 256, 0, stream>>>(loop_w, wt, agg, aggN);
    k1_tangent<<<(n + 7) / 8,        256, 0, stream>>>(h_hyper, xt, xtb, radius, n);
    k2_edges  <<<(e + 7) / 8,        256, 0, stream>>>(xt, radius, weight, src, dst,
                                                       etype, agg, e);
    k3_gemm   <<<((n / 16) * 16 + 7) / 8, 256, 0, stream>>>(xtb, wt, normv, agg, n);
    k4_expmap <<<(n + 7) / 8,        256, 0, stream>>>(agg, out, n);
    (void)out_size; (void)n_in;
}
